// GraphormerEmbeddingPP_45054206935227
// MI455X (gfx1250) — matrix operand order repaired and checked by reading, not compiled, not run on hardware
//
#include <hip/hip_runtime.h>
#include <hip/hip_bf16.h>
#include <math.h>

typedef __attribute__((ext_vector_type(16))) _Float16 v16h;
typedef __attribute__((ext_vector_type(8)))  float    v8f;

#define B_    2
#define N_    256
#define NP1_  257
#define D_    768
#define H_    160
#define K3_   128
#define MD_   5
#define NSQ_  66049            // 257*257

// output layout (floats): x | padding_mask | g | input_ids | llm_mask
#define XOFF_    0
#define PMOFF_   394752        // 257*2*768
#define GOFF_    395266        // + 2*257
#define IDSOFF_  21530946      // + 2*160*257*257
#define LLMOFF_  21530978

__device__ __forceinline__ v8f wmma_f16(v16h a, v16h b, v8f c) {
  return __builtin_amdgcn_wmma_f32_16x16x32_f16(false, a, false, b, (short)0, c,
                                                false, false);
}

// ---------------------------------------------------------------------------
// Kernel: inner region of g = 2*attn + spatial_bias + edge_bias  (WMMA)
// edge_bias[m,k] = (sum_d sum_h edge_emb[eidx[m,d]][h] * edge_dis[d][h][k]) / sp
// ---------------------------------------------------------------------------
__global__ __launch_bounds__(256) void k_inner_g(
    const float* __restrict__ attn,         // [B,257,257]
    const float* __restrict__ spatial_emb,  // [512,160]
    const float* __restrict__ edge_emb,     // [1537,160]
    const float* __restrict__ edge_dis,     // [5,160,160]
    const int*   __restrict__ spatial_pos,  // [B,256,256]
    const int*   __restrict__ edge_input,   // [B,256,256,5,1]
    float* __restrict__ g)                  // [B,160,257,257]
{
  __shared__ __align__(32) _Float16 bT[H_ * H_];   // edge_dis[d] transposed: bT[k*160+h]

  const int lane = threadIdx.x & 31;
  const int wv   = threadIdx.x >> 5;
  const int hi   = lane >> 4;     // lane group (0: lanes 0-15, 1: 16-31)
  const int ln   = lane & 15;
  const int M0w  = blockIdx.x * 128 + wv * 16;
  const int b    = M0w >> 16;
  const int i    = (M0w >> 8) & 255;
  const int jb   = M0w & 255;     // 16-aligned

  v8f c[10];
  #pragma unroll
  for (int t = 0; t < 10; ++t) c[t] = {};

  for (int d = 0; d < MD_; ++d) {
    __syncthreads();
    const float* src = edge_dis + d * H_ * H_;
    for (int t = threadIdx.x; t < H_ * H_; t += 256) {
      int h = t / H_, k = t % H_;
      bT[k * H_ + h] = (_Float16)src[t];                 // transpose while converting
    }
    __syncthreads();

    // A tile: rows = j = jb..jb+15, columns = h (K dim, 160)
    const int j  = jb + ln;
    const int ei = edge_input[((b * N_ + i) * N_ + j) * MD_ + d];
    const float* arow = edge_emb + ei * H_;

    for (int kt = 0; kt < 5; ++kt) {
      v16h a;
      const float* p = arow + kt * 32 + hi * 16;
      #pragma unroll
      for (int v = 0; v < 8; ++v) {
        int K = 2 * v;
        float2 f2 = *(const float2*)(p + K);
        a[2 * v]     = (_Float16)f2.x;
        a[2 * v + 1] = (_Float16)f2.y;
      }
      #pragma unroll
      for (int nt = 0; nt < 10; ++nt) {
        v16h bb = *(const v16h*)(bT + (nt * 16 + ln) * H_ + kt * 32 + hi * 16);
        c[nt] = wmma_f16(a, bb, c[nt]);
      }
    }
  }

  // epilogue: /sp + spatial_emb + 2*attn, write g inner region
  int   rowoff[8], spbase[8];
  float spfac[8], a2[8];
  #pragma unroll
  for (int r = 0; r < 8; ++r) {
    int m  = r + 8 * hi;
    int j  = jb + m;
    int sp0 = spatial_pos[(b * N_ + i) * N_ + j];
    int s = (sp0 == 0) ? 1 : sp0;
    if (s > 1) s -= 1;
    s = (s < 0) ? 0 : ((s > MD_) ? MD_ : s);
    spfac[r]  = 1.0f / (float)s;
    spbase[r] = sp0 * H_;
    a2[r]     = 2.0f * attn[b * NSQ_ + (i + 1) * NP1_ + (j + 1)];
    rowoff[r] = b * H_ * NSQ_ + (i + 1) * NP1_ + (j + 1);
  }
  #pragma unroll
  for (int nt = 0; nt < 10; ++nt) {
    int h = nt * 16 + ln;
    #pragma unroll
    for (int r = 0; r < 8; ++r) {
      float val = c[nt][r] * spfac[r] + spatial_emb[spbase[r] + h] + a2[r];
      g[rowoff[r] + h * NSQ_] = val;
    }
  }
}

// ---------------------------------------------------------------------------
// Kernel: border of g (row ii==0 and column jj==0): 2*attn + vdist term
// ---------------------------------------------------------------------------
__global__ __launch_bounds__(288) void k_border_g(
    const float* __restrict__ attn, const float* __restrict__ vdist,
    float* __restrict__ g)
{
  int t = threadIdx.x;
  if (t >= NP1_) return;
  int b = blockIdx.x / H_;
  int h = blockIdx.x % H_;
  float tv = vdist[h];
  float* gb = g + (b * H_ + h) * NSQ_;
  const float* ab = attn + b * NSQ_;
  gb[t] = 2.0f * ab[t] + tv;                       // ii == 0 row (all jj)
  if (t >= 1)
    gb[t * NP1_] = 2.0f * ab[t * NP1_] + tv;       // jj == 0 col (ii >= 1)
}

// ---------------------------------------------------------------------------
// Kernel: ef_sum[b,i,k] = sum_j (!pad[j]) gauss(b,i,j,k)
// ---------------------------------------------------------------------------
__global__ __launch_bounds__(128) void k_efsum(
    const float* __restrict__ pos, const float* __restrict__ gbf_means,
    const float* __restrict__ gbf_stds, const float* __restrict__ gbf_mul,
    const float* __restrict__ gbf_bias, const int* __restrict__ x0,
    const int* __restrict__ ntedge, float* __restrict__ efsum)
{
  __shared__ float distS[N_], mulS[N_], biaS[N_];
  __shared__ int   padS[N_];
  int b = blockIdx.x >> 8;
  int i = blockIdx.x & 255;
  float px = pos[(b * N_ + i) * 3 + 0];
  float py = pos[(b * N_ + i) * 3 + 1];
  float pz = pos[(b * N_ + i) * 3 + 2];
  for (int j = threadIdx.x; j < N_; j += 128) {
    float dx = px - pos[(b * N_ + j) * 3 + 0];
    float dy = py - pos[(b * N_ + j) * 3 + 1];
    float dz = pz - pos[(b * N_ + j) * 3 + 2];
    distS[j] = sqrtf(dx * dx + dy * dy + dz * dz);
    int e0 = ntedge[((b * N_ + i) * N_ + j) * 2 + 0];
    int e1 = ntedge[((b * N_ + i) * N_ + j) * 2 + 1];
    mulS[j] = gbf_mul[e0] + gbf_mul[e1];
    biaS[j] = gbf_bias[e0] + gbf_bias[e1];
    padS[j] = (x0[(b * N_ + j) * 9] == 0);
  }
  __syncthreads();
  int k = threadIdx.x;
  float mean = gbf_means[k];
  float sd   = fabsf(gbf_stds[k]) + 1e-5f;
  float rinv = 1.0f / sd;
  float rnrm = 0.3989422804014327f / sd;
  float acc = 0.0f;
  for (int j = 0; j < N_; ++j) {
    if (!padS[j]) {
      float t = (mulS[j] * distS[j] + biaS[j] - mean) * rinv;
      acc += expf(-0.5f * t * t) * rnrm;
    }
  }
  efsum[(b * N_ + i) * K3_ + k] = acc;
}

// ---------------------------------------------------------------------------
// Kernel: bias3d = w2( gelu( w1(gauss) ) ) accumulated into g (WMMA x2)
// ---------------------------------------------------------------------------
__global__ __launch_bounds__(256) void k_bias3d(
    const float* __restrict__ pos, const float* __restrict__ gbf_means,
    const float* __restrict__ gbf_stds, const float* __restrict__ gbf_mul,
    const float* __restrict__ gbf_bias, const float* __restrict__ w1,
    const float* __restrict__ b1, const float* __restrict__ w2,
    const float* __restrict__ b2, const int* __restrict__ x0,
    const int* __restrict__ ntedge, const unsigned char* __restrict__ mask3d,
    float* __restrict__ g)
{
  __shared__ __align__(32) _Float16 w1T[K3_ * K3_];     // 32 KB  w1T[n*128+k]
  __shared__ __align__(32) _Float16 w2T[H_ * K3_];      // 40 KB  w2T[n*128+k]
  __shared__ __align__(32) _Float16 hidS[8 * 16 * K3_]; // 32 KB  per-wave 16x128
  __shared__ float meanS[K3_], rinvS[K3_], rnrmS[K3_], b1S[K3_], b2S[H_];

  const int lane = threadIdx.x & 31;
  const int wv   = threadIdx.x >> 5;
  const int hi   = lane >> 4;
  const int ln   = lane & 15;
  const int M0w  = blockIdx.x * 128 + wv * 16;
  const int b    = M0w >> 16;
  const int i    = (M0w >> 8) & 255;
  const int jb   = M0w & 255;

  for (int t = threadIdx.x; t < K3_ * K3_; t += 256)
    w1T[(t & 127) * K3_ + (t >> 7)] = (_Float16)w1[t];          // t = k*128+n
  for (int t = threadIdx.x; t < K3_ * H_; t += 256) {
    int k = t / H_, n = t % H_;
    w2T[n * K3_ + k] = (_Float16)w2[t];
  }
  for (int t = threadIdx.x; t < K3_; t += 256) {
    float sd = fabsf(gbf_stds[t]) + 1e-5f;
    meanS[t] = gbf_means[t];
    rinvS[t] = 1.0f / sd;
    rnrmS[t] = 0.3989422804014327f / sd;
    b1S[t]   = b1[t];
  }
  for (int t = threadIdx.x; t < H_; t += 256) b2S[t] = b2[t];
  __syncthreads();

  // per-row scalars (row m = ln); lanes l and l+16 redundantly compute same row
  const int j = jb + ln;
  float dx = pos[(b * N_ + i) * 3 + 0] - pos[(b * N_ + j) * 3 + 0];
  float dy = pos[(b * N_ + i) * 3 + 1] - pos[(b * N_ + j) * 3 + 1];
  float dz = pos[(b * N_ + i) * 3 + 2] - pos[(b * N_ + j) * 3 + 2];
  float dist = sqrtf(dx * dx + dy * dy + dz * dz);
  int e0 = ntedge[((b * N_ + i) * N_ + j) * 2 + 0];
  int e1 = ntedge[((b * N_ + i) * N_ + j) * 2 + 1];
  float xg = (gbf_mul[e0] + gbf_mul[e1]) * dist + gbf_bias[e0] + gbf_bias[e1];

  // GEMM1: gauss (16x128) @ w1 (128x128)
  v8f c1[8];
  #pragma unroll
  for (int t = 0; t < 8; ++t) c1[t] = {};
  for (int kt = 0; kt < 4; ++kt) {
    v16h a;
    #pragma unroll
    for (int v = 0; v < 8; ++v) {
      int K = kt * 32 + hi * 16 + 2 * v;
      float t0 = (xg - meanS[K]) * rinvS[K];
      float t1 = (xg - meanS[K + 1]) * rinvS[K + 1];
      a[2 * v]     = (_Float16)(expf(-0.5f * t0 * t0) * rnrmS[K]);
      a[2 * v + 1] = (_Float16)(expf(-0.5f * t1 * t1) * rnrmS[K + 1]);
    }
    #pragma unroll
    for (int nt = 0; nt < 8; ++nt) {
      v16h bb = *(const v16h*)(w1T + (nt * 16 + ln) * K3_ + kt * 32 + hi * 16);
      c1[nt] = wmma_f16(a, bb, c1[nt]);
    }
  }

  // gelu(exact) + transpose C-layout -> row-major hid tile in LDS
  _Float16* hw = hidS + wv * 16 * K3_;
  #pragma unroll
  for (int nt = 0; nt < 8; ++nt) {
    int k = nt * 16 + ln;
    float bk = b1S[k];
    #pragma unroll
    for (int r = 0; r < 8; ++r) {
      int m = r + 8 * hi;
      float x = c1[nt][r] + bk;
      float gel = 0.5f * x * (1.0f + erff(x * 0.70710678118654752f));
      hw[m * K3_ + k] = (_Float16)gel;
    }
  }
  __syncthreads();

  // GEMM2: hid (16x128) @ w2 (128x160)
  v8f c2[10];
  #pragma unroll
  for (int t = 0; t < 10; ++t) c2[t] = {};
  for (int kt = 0; kt < 4; ++kt) {
    v16h a;
    #pragma unroll
    for (int v = 0; v < 8; ++v) {
      int K = kt * 32 + hi * 16 + 2 * v;
      a[2 * v]     = hw[ln * K3_ + K];
      a[2 * v + 1] = hw[ln * K3_ + K + 1];
    }
    #pragma unroll
    for (int nt = 0; nt < 10; ++nt) {
      v16h bb = *(const v16h*)(w2T + (nt * 16 + ln) * K3_ + kt * 32 + hi * 16);
      c2[nt] = wmma_f16(a, bb, c2[nt]);
    }
  }

  // epilogue: +b2, pad -> -inf, mask3d -> skip, accumulate into g inner region
  bool m3 = (mask3d[b] != 0);
  int  rowoff[8];
  bool padr[8];
  #pragma unroll
  for (int r = 0; r < 8; ++r) {
    int m = r + 8 * hi, jj = jb + m;
    padr[r]   = (x0[(b * N_ + jj) * 9] == 0);
    rowoff[r] = b * H_ * NSQ_ + (i + 1) * NP1_ + (jj + 1);
  }
  if (!m3) {
    #pragma unroll
    for (int nt = 0; nt < 10; ++nt) {
      int h = nt * 16 + ln;
      float bb2 = b2S[h];
      #pragma unroll
      for (int r = 0; r < 8; ++r) {
        float val = padr[r] ? -__builtin_inff() : (c2[nt][r] + bb2);
        int idx = rowoff[r] + h * NSQ_;
        g[idx] += val;
      }
    }
  }
}

// ---------------------------------------------------------------------------
// Kernel: x = LN(node_feat + 0.01*merged), transposed to [N+1, B, D]
// ---------------------------------------------------------------------------
__global__ __launch_bounds__(256) void k_xout(
    const float* __restrict__ atom_emb, const float* __restrict__ in_emb,
    const float* __restrict__ out_emb, const float* __restrict__ gtok,
    const float* __restrict__ epw, const float* __restrict__ epb,
    const float* __restrict__ lnw, const float* __restrict__ lnb,
    const int* __restrict__ x0, const int* __restrict__ indeg,
    const int* __restrict__ outdeg, const unsigned char* __restrict__ mask3d,
    const float* __restrict__ efsum, float* __restrict__ xout)
{
  __shared__ float efsS[K3_];
  __shared__ float red[256];
  int b  = blockIdx.x / NP1_;
  int ii = blockIdx.x % NP1_;
  float v[3];

  if (ii == 0) {
    #pragma unroll
    for (int p = 0; p < 3; ++p) v[p] = gtok[threadIdx.x + 256 * p];
  } else {
    int i = ii - 1;
    for (int t = threadIdx.x; t < K3_; t += 256)
      efsS[t] = efsum[(b * N_ + i) * K3_ + t];
    __syncthreads();
    int id[9];
    #pragma unroll
    for (int t = 0; t < 9; ++t) id[t] = x0[(b * N_ + i) * 9 + t];
    int di  = indeg[b * N_ + i];
    int dou = outdeg[b * N_ + i];
    bool m3 = (mask3d[b] != 0);
    #pragma unroll
    for (int p = 0; p < 3; ++p) {
      int d = threadIdx.x + 256 * p;
      float s = in_emb[di * D_ + d] + out_emb[dou * D_ + d];
      for (int t = 0; t < 9; ++t) s += atom_emb[id[t] * D_ + d];
      if (!m3) {
        float mg = epb[d];
        for (int k = 0; k < K3_; ++k) mg += efsS[k] * epw[k * D_ + d];
        s += 0.01f * mg;
      }
      v[p] = s;
    }
  }

  float ps = 0.f, pss = 0.f;
  #pragma unroll
  for (int p = 0; p < 3; ++p) { ps += v[p]; pss += v[p] * v[p]; }
  __syncthreads();
  red[threadIdx.x] = ps;
  __syncthreads();
  for (int s = 128; s > 0; s >>= 1) {
    if (threadIdx.x < s) red[threadIdx.x] += red[threadIdx.x + s];
    __syncthreads();
  }
  float mu = red[0] * (1.0f / 768.0f);
  __syncthreads();
  red[threadIdx.x] = pss;
  __syncthreads();
  for (int s = 128; s > 0; s >>= 1) {
    if (threadIdx.x < s) red[threadIdx.x] += red[threadIdx.x + s];
    __syncthreads();
  }
  float var = red[0] * (1.0f / 768.0f) - mu * mu;
  float rs  = rsqrtf(var + 1e-5f);
  #pragma unroll
  for (int p = 0; p < 3; ++p) {
    int d = threadIdx.x + 256 * p;
    xout[(ii * B_ + b) * D_ + d] = (v[p] - mu) * rs * lnw[d] + lnb[d];
  }
}

// ---------------------------------------------------------------------------
// Kernel: padding_mask + input_ids / llm_mask passthrough (as float)
// ---------------------------------------------------------------------------
__global__ __launch_bounds__(640) void k_misc(
    const int* __restrict__ x0, const int* __restrict__ input_ids,
    const unsigned char* __restrict__ llm, float* __restrict__ out)
{
  int t = threadIdx.x;
  if (t < 514) {
    int b = t / NP1_, jj = t % NP1_;
    float val = (jj > 0 && x0[(b * N_ + (jj - 1)) * 9] == 0) ? 1.0f : 0.0f;
    out[PMOFF_ + t] = val;
  } else if (t < 546) {
    int q = t - 514;
    out[IDSOFF_ + q] = (float)input_ids[q];
  } else if (t < 578) {
    int q = t - 546;
    out[LLMOFF_ + q] = llm[q] ? 1.0f : 0.0f;
  }
}

// ---------------------------------------------------------------------------
extern "C" void kernel_launch(void* const* d_in, const int* in_sizes, int n_in,
                              void* d_out, int out_size, void* d_ws, size_t ws_size,
                              hipStream_t stream) {
  const float* attn        = (const float*)d_in[0];
  const float* pos         = (const float*)d_in[1];
  const float* atom_emb    = (const float*)d_in[2];
  const float* in_emb      = (const float*)d_in[3];
  const float* out_emb     = (const float*)d_in[4];
  const float* gtok        = (const float*)d_in[5];
  const float* spatial_emb = (const float*)d_in[6];
  const float* vdist       = (const float*)d_in[7];
  const float* edge_emb    = (const float*)d_in[8];
  const float* edge_dis    = (const float*)d_in[9];
  const float* gbf_means   = (const float*)d_in[10];
  const float* gbf_stds    = (const float*)d_in[11];
  const float* gbf_mul     = (const float*)d_in[12];
  const float* gbf_bias    = (const float*)d_in[13];
  const float* gbf_w1      = (const float*)d_in[14];
  const float* gbf_b1      = (const float*)d_in[15];
  const float* gbf_w2      = (const float*)d_in[16];
  const float* gbf_b2      = (const float*)d_in[17];
  const float* edge_proj_w = (const float*)d_in[18];
  const float* edge_proj_b = (const float*)d_in[19];
  const float* ln_w        = (const float*)d_in[20];
  const float* ln_b        = (const float*)d_in[21];
  const int*   x0          = (const int*)d_in[22];
  const int*   in_degree   = (const int*)d_in[23];
  const int*   out_degree  = (const int*)d_in[24];
  const int*   spatial_pos = (const int*)d_in[25];
  const int*   edge_input  = (const int*)d_in[26];
  const int*   ntedge      = (const int*)d_in[27];
  const unsigned char* m3  = (const unsigned char*)d_in[28];
  const int*   input_ids   = (const int*)d_in[29];
  const unsigned char* llm = (const unsigned char*)d_in[30];

  float* out   = (float*)d_out;
  float* g     = out + GOFF_;
  float* efsum = (float*)d_ws;               // B*N*128 f32 = 256 KB

  // 1) inner g = 2*attn + spatial + edge_bias  (WMMA head mix)
  k_inner_g<<<1024, 256, 0, stream>>>(attn, spatial_emb, edge_emb, edge_dis,
                                      spatial_pos, edge_input, g);
  // 2) g borders
  k_border_g<<<B_ * H_, 288, 0, stream>>>(attn, vdist, g);
  // 3) ef_sum for merged
  k_efsum<<<B_ * N_, 128, 0, stream>>>(pos, gbf_means, gbf_stds, gbf_mul,
                                       gbf_bias, x0, ntedge, efsum);
  // 4) bias3d MLP (WMMA x2), accumulate into g
  k_bias3d<<<1024, 256, 0, stream>>>(pos, gbf_means, gbf_stds, gbf_mul,
                                     gbf_bias, gbf_w1, gbf_b1, gbf_w2, gbf_b2,
                                     x0, ntedge, m3, g);
  // 5) x output (node features + merged + LayerNorm, transposed)
  k_xout<<<B_ * NP1_, 256, 0, stream>>>(atom_emb, in_emb, out_emb, gtok,
                                        edge_proj_w, edge_proj_b, ln_w, ln_b,
                                        x0, in_degree, out_degree, m3, efsum,
                                        out + XOFF_);
  // 6) padding mask + passthroughs
  k_misc<<<1, 640, 0, stream>>>(x0, input_ids, llm, out);
}
